// RecordMemory_12292196402102
// MI455X (gfx1250) — matrix operand order repaired and checked by reading, not compiled, not run on hardware
//
#include <hip/hip_runtime.h>
#include <hip/hip_bf16.h>
#include <math.h>

typedef __attribute__((ext_vector_type(16))) __bf16 v16bf;
typedef __attribute__((ext_vector_type(8)))  __bf16 v8bf;
typedef __attribute__((ext_vector_type(8)))  float  v8f;
typedef __attribute__((ext_vector_type(8)))  float  v8fv;

#define B_ROWS   512
#define DIM      256
#define NPROXY   65536
#define NCLASS   8192
#define INV_TEMP 20.0f
#define BG_KNN   50
#define NEG_FILL (-10000.0f)
#define SLICE    1024
#define PSLICES  (NPROXY / SLICE)   /* 64 */
#define CSLICES  (NCLASS / SLICE)   /* 8  */
#define RBLK     16
#define NRB      (B_ROWS / RBLK)    /* 32 */

// ---------------- fragment loaders ----------------
// A-operand 16x32 bf16: lane m holds K = kg*16+{0..15} (contiguous)
// B-operand 32x16 bf16: lane n holds K = kg*16+{0..15} (contiguous)

__device__ __forceinline__ v16bf cat8(v8bf x, v8bf y) {
    return __builtin_shufflevector(x, y, 0, 1, 2, 3, 4, 5, 6, 7,
                                   8, 9, 10, 11, 12, 13, 14, 15);
}

__device__ __forceinline__ v16bf ldfragA_f32(const float* __restrict__ rp, int kbase, int kg) {
    v8fv x = *(const v8fv*)(rp + kbase + kg * 16);
    v8fv y = *(const v8fv*)(rp + kbase + kg * 16 + 8);
    return cat8(__builtin_convertvector(x, v8bf), __builtin_convertvector(y, v8bf));
}
__device__ __forceinline__ v16bf ldfragB_f32(const float* __restrict__ rp, int kbase, int kg) {
    v8fv x = *(const v8fv*)(rp + kbase + kg * 16);
    v8fv y = *(const v8fv*)(rp + kbase + kg * 16 + 8);
    return cat8(__builtin_convertvector(x, v8bf), __builtin_convertvector(y, v8bf));
}
__device__ __forceinline__ v16bf ldfragA_bf(const __bf16* __restrict__ rp, int kbase, int kg) {
    v8bf x = *(const v8bf*)(rp + kbase + kg * 16);
    v8bf y = *(const v8bf*)(rp + kbase + kg * 16 + 8);
    return cat8(x, y);
}
__device__ __forceinline__ v16bf ldfragB_bf(const __bf16* __restrict__ rp, int kbase, int kg) {
    return *(const v16bf*)(rp + kbase + kg * 16);
}

__device__ __forceinline__ v8f wmma_bf16(v16bf a, v16bf b, v8f c) {
    return __builtin_amdgcn_wmma_f32_16x16x32_bf16(false, a, false, b, (short)0, c,
                                                   false, false);
}

// ---------------------------------------------------------------------------
// Kernel 0: f32 -> bf16 bulk conversion (8 elements / thread, b128 in/out)
// ---------------------------------------------------------------------------
__launch_bounds__(256)
__global__ void k_cvt(const float* __restrict__ src, __bf16* __restrict__ dst, int n8) {
    int i = blockIdx.x * blockDim.x + threadIdx.x;
    if (i < n8) {
        v8fv x = *(const v8fv*)(src + (size_t)i * 8);
        *(v8bf*)(dst + (size_t)i * 8) = __builtin_convertvector(x, v8bf);
    }
}

// ---------------------------------------------------------------------------
// Kernel 1: fused WMMA GEMM (inputs @ proxy_centers^T)/TEMP + positive stats +
//           per-slice top-50 hard negatives. One WG = 16 rows x 1024 cols.
// ---------------------------------------------------------------------------
template <bool PRE>
__launch_bounds__(256)
__global__ void k_proxy(const float*  __restrict__ inputsF,
                        const float*  __restrict__ proxyF,
                        const __bf16* __restrict__ inputsB,
                        const __bf16* __restrict__ proxyB,
                        const int*    __restrict__ labels,
                        const int*    __restrict__ proxy2label,
                        float* __restrict__ posPart,
                        float* __restrict__ negCand) {
    extern __shared__ float negbuf[];            // [RBLK][SLICE] = 64 KB
    __shared__ float posS[RBLK], posC[RBLK], posE[RBLK];

    const int rb   = blockIdx.x % NRB;
    const int cs   = blockIdx.x / NRB;
    const int tid  = threadIdx.x;
    const int wave = tid >> 5;
    const int lane = tid & 31;
    const int l16  = lane & 15;
    const int kg   = lane >> 4;

    if (tid < RBLK) { posS[tid] = 0.0f; posC[tid] = 0.0f; posE[tid] = 0.0f; }
    __syncthreads();

    // loop-invariant labels for this lane's 8 output rows (registers, not LDS)
    int labR[8];
#pragma unroll
    for (int r = 0; r < 8; ++r) labR[r] = labels[rb * RBLK + r + 8 * kg];

    v16bf a[8];
#pragma unroll
    for (int kk = 0; kk < 8; ++kk) {
        if constexpr (PRE)
            a[kk] = ldfragA_bf(inputsB + (size_t)(rb * RBLK + l16) * DIM, kk * 32, kg);
        else
            a[kk] = ldfragA_f32(inputsF + (size_t)(rb * RBLK + l16) * DIM, kk * 32, kg);
    }

    for (int it = 0; it < SLICE / 128; ++it) {   // 8 iterations, 128 cols each
        const int colLocal = it * 128 + wave * 16 + l16;
        const int col      = cs * SLICE + colLocal;
        if (it + 1 < SLICE / 128) {
            if constexpr (PRE)
                __builtin_prefetch(proxyB + (size_t)(col + 128) * DIM, 0, 1);
            else
                __builtin_prefetch(proxyF + (size_t)(col + 128) * DIM, 0, 1);
        }

        v8f c = {0.f, 0.f, 0.f, 0.f, 0.f, 0.f, 0.f, 0.f};
#pragma unroll
        for (int kk = 0; kk < 8; ++kk) {
            v16bf bf;
            if constexpr (PRE)
                bf = ldfragB_bf(proxyB + (size_t)col * DIM, kk * 32, kg);
            else
                bf = ldfragB_f32(proxyF + (size_t)col * DIM, kk * 32, kg);
            c = wmma_bf16(a[kk], bf, c);
        }

        const int p2l = proxy2label[col];
#pragma unroll
        for (int r = 0; r < 8; ++r) {
            const int row = r + 8 * kg;
            float s = c[r] * INV_TEMP;
            bool pos = (labR[r] == p2l);
            negbuf[row * SLICE + colLocal] = pos ? NEG_FILL : s;
            if (pos) {
                atomicAdd(&posS[row], s);
                atomicAdd(&posC[row], 1.0f);
                atomicAdd(&posE[row], __expf(s));
            }
        }
    }
    __syncthreads();

    if (tid < RBLK) {
        float* pp = posPart + ((size_t)(rb * RBLK + tid) * PSLICES + cs) * 3;
        pp[0] = posS[tid]; pp[1] = posC[tid]; pp[2] = posE[tid];
    }

    // Per-wave top-50 extraction: wave w owns rows 2w, 2w+1
    volatile float* vb = negbuf;
    for (int rr = 0; rr < 2; ++rr) {
        const int row = wave * 2 + rr;
        float* outc = negCand + ((size_t)(rb * RBLK + row) * PSLICES + cs) * BG_KNN;
        for (int t = 0; t < BG_KNN; ++t) {
            float val = -3.0e38f; int idx = lane;
            for (int ci = lane; ci < SLICE; ci += 32) {
                float v = vb[row * SLICE + ci];
                if (v > val) { val = v; idx = ci; }
            }
#pragma unroll
            for (int off = 16; off >= 1; off >>= 1) {
                float ov = __shfl_xor(val, off, 32);
                int   oi = __shfl_xor(idx, off, 32);
                if (ov > val) { val = ov; idx = oi; }
            }
            if (lane == 0) {
                outc[t] = val;
                vb[row * SLICE + idx] = -3.0e38f;
            }
        }
    }
}

// ---------------------------------------------------------------------------
// Kernel 2: fused WMMA GEMM (inputs @ class_centers^T)/TEMP: per-row sum(exp)
//           and target logit partials.
// ---------------------------------------------------------------------------
template <bool PRE>
__launch_bounds__(256)
__global__ void k_class(const float*  __restrict__ inputsF,
                        const float*  __restrict__ classF,
                        const __bf16* __restrict__ inputsB,
                        const __bf16* __restrict__ classB,
                        const int*    __restrict__ classes,
                        float* __restrict__ clsPart) {
    __shared__ float eS[RBLK], tS[RBLK];

    const int rb   = blockIdx.x % NRB;
    const int cs   = blockIdx.x / NRB;
    const int tid  = threadIdx.x;
    const int wave = tid >> 5;
    const int lane = tid & 31;
    const int l16  = lane & 15;
    const int kg   = lane >> 4;

    if (tid < RBLK) { eS[tid] = 0.f; tS[tid] = 0.f; }
    __syncthreads();

    // loop-invariant target class ids for this lane's 8 output rows
    int clsR[8];
#pragma unroll
    for (int r = 0; r < 8; ++r) clsR[r] = classes[rb * RBLK + r + 8 * kg];

    v16bf a[8];
#pragma unroll
    for (int kk = 0; kk < 8; ++kk) {
        if constexpr (PRE)
            a[kk] = ldfragA_bf(inputsB + (size_t)(rb * RBLK + l16) * DIM, kk * 32, kg);
        else
            a[kk] = ldfragA_f32(inputsF + (size_t)(rb * RBLK + l16) * DIM, kk * 32, kg);
    }

    float es[8] = {0.f,0.f,0.f,0.f,0.f,0.f,0.f,0.f};
    float ts[8] = {0.f,0.f,0.f,0.f,0.f,0.f,0.f,0.f};

    for (int it = 0; it < SLICE / 128; ++it) {
        const int col = cs * SLICE + it * 128 + wave * 16 + l16;
        if (it + 1 < SLICE / 128) {
            if constexpr (PRE)
                __builtin_prefetch(classB + (size_t)(col + 128) * DIM, 0, 1);
            else
                __builtin_prefetch(classF + (size_t)(col + 128) * DIM, 0, 1);
        }

        v8f c = {0.f, 0.f, 0.f, 0.f, 0.f, 0.f, 0.f, 0.f};
#pragma unroll
        for (int kk = 0; kk < 8; ++kk) {
            v16bf bf;
            if constexpr (PRE)
                bf = ldfragB_bf(classB + (size_t)col * DIM, kk * 32, kg);
            else
                bf = ldfragB_f32(classF + (size_t)col * DIM, kk * 32, kg);
            c = wmma_bf16(a[kk], bf, c);
        }
#pragma unroll
        for (int r = 0; r < 8; ++r) {
            float s = c[r] * INV_TEMP;
            es[r] += __expf(s);
            if (col == clsR[r]) ts[r] += s;
        }
    }

#pragma unroll
    for (int r = 0; r < 8; ++r) {
        float e = es[r], t = ts[r];
#pragma unroll
        for (int off = 1; off <= 8; off <<= 1) {
            e += __shfl_xor(e, off, 32);
            t += __shfl_xor(t, off, 32);
        }
        if (l16 == 0) {
            const int row = r + 8 * kg;
            atomicAdd(&eS[row], e);
            atomicAdd(&tS[row], t);
        }
    }
    __syncthreads();
    if (tid < RBLK) {
        float* cp = clsPart + ((size_t)(rb * RBLK + tid) * CSLICES + cs) * 2;
        cp[0] = eS[tid]; cp[1] = tS[tid];
    }
}

// ---------------------------------------------------------------------------
// Kernel 3: per-row merge of proxy partials: global top-50 of 64x50 candidates
//           + combine positive stats -> lse - mean_pos
// ---------------------------------------------------------------------------
__launch_bounds__(32)
__global__ void k_merge_proxy(const float* __restrict__ posPart,
                              const float* __restrict__ negCand,
                              float* __restrict__ rowLossP) {
    __shared__ float cand[PSLICES * BG_KNN];     // 3200 floats
    const int row  = blockIdx.x;
    const int lane = threadIdx.x;

    for (int i = lane; i < PSLICES * BG_KNN; i += 32)
        cand[i] = negCand[(size_t)row * PSLICES * BG_KNN + i];

    float ps = 0.f, pc = 0.f, pe = 0.f;
    for (int i = lane; i < PSLICES; i += 32) {
        const float* pp = posPart + ((size_t)row * PSLICES + i) * 3;
        ps += pp[0]; pc += pp[1]; pe += pp[2];
    }
#pragma unroll
    for (int off = 16; off >= 1; off >>= 1) {
        ps += __shfl_xor(ps, off, 32);
        pc += __shfl_xor(pc, off, 32);
        pe += __shfl_xor(pe, off, 32);
    }
    __syncthreads();

    volatile float* vc = cand;
    float negE = 0.0f;
    for (int t = 0; t < BG_KNN; ++t) {
        float val = -3.0e38f; int idx = lane;
        for (int i = lane; i < PSLICES * BG_KNN; i += 32) {
            float v = vc[i];
            if (v > val) { val = v; idx = i; }
        }
#pragma unroll
        for (int off = 16; off >= 1; off >>= 1) {
            float ov = __shfl_xor(val, off, 32);
            int   oi = __shfl_xor(idx, off, 32);
            if (ov > val) { val = ov; idx = oi; }
        }
        if (lane == 0) {
            vc[idx] = -3.0e38f;
            negE += __expf(val);
        }
    }
    if (lane == 0) {
        float lse = __logf(pe + negE);           // |scores| <= 20 -> exp in f32 range
        rowLossP[row] = lse - ps / pc;
    }
}

// ---------------------------------------------------------------------------
// Kernel 4: per-row class CE merge
// ---------------------------------------------------------------------------
__global__ void k_merge_class(const float* __restrict__ clsPart,
                              float* __restrict__ rowLossC) {
    const int row = blockIdx.x * blockDim.x + threadIdx.x;
    if (row < B_ROWS) {
        float e = 0.f, t = 0.f;
        for (int s = 0; s < CSLICES; ++s) {
            const float* cp = clsPart + ((size_t)row * CSLICES + s) * 2;
            e += cp[0]; t += cp[1];
        }
        rowLossC[row] = __logf(e) - t;
    }
}

// ---------------------------------------------------------------------------
// Kernel 5: deterministic final reduction to scalar loss
// ---------------------------------------------------------------------------
__launch_bounds__(256)
__global__ void k_final(const float* __restrict__ rowP,
                        const float* __restrict__ rowC,
                        float* __restrict__ out) {
    __shared__ float red[256];
    const int tid = threadIdx.x;
    float s = 0.f;
    for (int i = tid; i < B_ROWS; i += 256) s += rowP[i] + rowC[i];
    red[tid] = s;
    __syncthreads();
    for (int off = 128; off >= 1; off >>= 1) {
        if (tid < off) red[tid] += red[tid + off];
        __syncthreads();
    }
    if (tid == 0) out[0] = red[0] * (1.0f / (float)B_ROWS);
}

extern "C" void kernel_launch(void* const* d_in, const int* in_sizes, int n_in,
                              void* d_out, int out_size, void* d_ws, size_t ws_size,
                              hipStream_t stream) {
    const float* inputs        = (const float*)d_in[0];
    const float* proxy_centers = (const float*)d_in[1];
    const float* class_centers = (const float*)d_in[2];
    /* d_in[3] = proxies (unused: labels already derived) */
    const int*   labels        = (const int*)d_in[4];
    const int*   classes       = (const int*)d_in[5];
    const int*   proxy2label   = (const int*)d_in[6];

    // float scratch region
    float* ws      = (float*)d_ws;
    float* posPart = ws;                                           // 512*64*3
    float* negCand = posPart + (size_t)B_ROWS * PSLICES * 3;       // 512*64*50
    float* clsPart = negCand + (size_t)B_ROWS * PSLICES * BG_KNN;  // 512*8*2
    float* rowP    = clsPart + (size_t)B_ROWS * CSLICES * 2;       // 512
    float* rowC    = rowP + B_ROWS;                                // 512
    const size_t floatBytes = (size_t)(rowC + B_ROWS - ws) * sizeof(float); // ~7 MB

    // bf16 scratch region (pre-converted operands), after float region
    __bf16* inputsB = (__bf16*)((char*)d_ws + floatBytes);
    __bf16* proxyB  = inputsB + (size_t)B_ROWS * DIM;
    __bf16* classB  = proxyB + (size_t)NPROXY * DIM;
    const size_t totalBytes = floatBytes +
        ((size_t)B_ROWS + NPROXY + NCLASS) * DIM * sizeof(__bf16); // ~45 MB

    const bool pre = (ws_size >= totalBytes);   // deterministic: ws_size is fixed

    if (pre) {
        const int nInp = (B_ROWS * DIM) / 8;    // 16384
        const int nPrx = (NPROXY * DIM) / 8;    // 2097152
        const int nCls = (NCLASS * DIM) / 8;    // 262144
        k_cvt<<<dim3((nInp + 255) / 256), dim3(256), 0, stream>>>(inputs, inputsB, nInp);
        k_cvt<<<dim3((nPrx + 255) / 256), dim3(256), 0, stream>>>(proxy_centers, proxyB, nPrx);
        k_cvt<<<dim3((nCls + 255) / 256), dim3(256), 0, stream>>>(class_centers, classB, nCls);
        k_proxy<true><<<dim3(NRB * PSLICES), dim3(256), RBLK * SLICE * sizeof(float), stream>>>(
            inputs, proxy_centers, inputsB, proxyB, labels, proxy2label, posPart, negCand);
        k_class<true><<<dim3(NRB * CSLICES), dim3(256), 0, stream>>>(
            inputs, class_centers, inputsB, classB, classes, clsPart);
    } else {
        k_proxy<false><<<dim3(NRB * PSLICES), dim3(256), RBLK * SLICE * sizeof(float), stream>>>(
            inputs, proxy_centers, inputsB, proxyB, labels, proxy2label, posPart, negCand);
        k_class<false><<<dim3(NRB * CSLICES), dim3(256), 0, stream>>>(
            inputs, class_centers, inputsB, classB, classes, clsPart);
    }

    k_merge_proxy<<<dim3(B_ROWS), dim3(32), 0, stream>>>(posPart, negCand, rowP);
    k_merge_class<<<dim3(2), dim3(256), 0, stream>>>(clsPart, rowC);
    k_final<<<dim3(1), dim3(256), 0, stream>>>(rowP, rowC, (float*)d_out);
}
